// QMeasurementC_15685220565667
// MI455X (gfx1250) — matrix operand order repaired and checked by reading, not compiled, not run on hardware
//
#include <hip/hip_runtime.h>

// ---------------------------------------------------------------------------
// QMeasurement: out[b,i] = sum_{j,l} x[b,j,l] * k[i,j] * conj(k[i,l])
// == GEMM [B=2048, P=16384] x [P, E=128] with M[i,p] = k[i,j]*conj(k[i,l]).
// Phase 1: build M (re/im) in bf16, pre-swizzled into WMMA B-fragment layout.
// Phase 2: split-K WMMA GEMM (bf16 in, f32 accum), partials to workspace.
// Phase 3: deterministic reduction over K-slices -> interleaved complex64 out.
// ---------------------------------------------------------------------------

typedef __bf16 bf16x16 __attribute__((ext_vector_type(16)));
typedef float  f32x8   __attribute__((ext_vector_type(8)));

#define E          128
#define P          16384                 // E*E (GEMM K dimension)
#define BATCH      2048
#define NSLICE     16                    // split-K factor
#define KSLICE     (P / NSLICE)          // 1024
#define OUT_FLOATS (BATCH * E * 2)       // 524288 floats (complex64 interleaved)

// ------------------------- Phase 1: projector fragments --------------------
// B-fragment layout for v_wmma_f32_16x16x32_bf16 (32x16 K-major B operand):
//   fragment id = (kChunk*8 + nChunk)*32 + lane, 16 contiguous bf16 per lane.
//   lanes 0-15 hold K {0..7,16..23}, lanes 16-31 hold K {8..15,24..31}.
//   N (the 'i' index) = nChunk*16 + (lane & 15).
__global__ void build_projector_frags(const float* __restrict__ kparam,
                                      __bf16* __restrict__ Br,
                                      __bf16* __restrict__ Bi) {
    int tid  = blockIdx.x * blockDim.x + threadIdx.x;   // 0 .. P*E-1
    int elem = tid & 15;
    int lane = (tid >> 4) & 31;
    int nc   = (tid >> 9) & 7;
    int kc   = tid >> 12;
    int kloc = ((lane < 16) ? 0 : 8) + ((elem < 8) ? elem : (8 + elem));
    int p    = kc * 32 + kloc;                          // 0 .. P-1
    int i    = nc * 16 + (lane & 15);                   // 0 .. E-1
    int j    = p >> 7;
    int l    = p & 127;

    float krj = kparam[(i * E + j) * 2 + 0];
    float kij = kparam[(i * E + j) * 2 + 1];
    float krl = kparam[(i * E + l) * 2 + 0];
    float kil = kparam[(i * E + l) * 2 + 1];
    // M[i,j,l] = k[i,j] * conj(k[i,l])
    float mr = krj * krl + kij * kil;
    float mi = kij * krl - krj * kil;
    Br[tid] = (__bf16)mr;
    Bi[tid] = (__bf16)mi;
}

// ------------------------------- helpers -----------------------------------
static __device__ inline bf16x16 cvt_frag(float4 a, float4 b, float4 c, float4 d) {
    bf16x16 r;
    r[0]  = (__bf16)a.x; r[1]  = (__bf16)a.y; r[2]  = (__bf16)a.z; r[3]  = (__bf16)a.w;
    r[4]  = (__bf16)b.x; r[5]  = (__bf16)b.y; r[6]  = (__bf16)b.z; r[7]  = (__bf16)b.w;
    r[8]  = (__bf16)c.x; r[9]  = (__bf16)c.y; r[10] = (__bf16)c.z; r[11] = (__bf16)c.w;
    r[12] = (__bf16)d.x; r[13] = (__bf16)d.y; r[14] = (__bf16)d.z; r[15] = (__bf16)d.w;
    return r;
}

// bf16 WMMA has no A/B negate modifier (ISA: NEG[1:0] must be 0) -> sign-flip
// the packed fragment with XOR.
static __device__ inline bf16x16 neg_frag(bf16x16 v) {
    union { bf16x16 v; unsigned u[8]; } t;
    t.v = v;
#pragma unroll
    for (int j = 0; j < 8; ++j) t.u[j] ^= 0x80008000u;
    return t.v;
}

// ------------------------- Phase 2: split-K WMMA GEMM -----------------------
// One wave = one 16-row batch block x all 8 N-tiles x one K-slice of 1024.
// 2048 waves total (128 row blocks x 16 slices), 8 waves per 256-thread block.
__global__ __launch_bounds__(256) void qmeas_gemm(
    const float*  __restrict__ Xr,
    const float*  __restrict__ Xi,
    const __bf16* __restrict__ Br,
    const __bf16* __restrict__ Bi,
    float*        __restrict__ partial) {
    const int lane     = threadIdx.x & 31;
    const int wid      = blockIdx.x * (blockDim.x >> 5) + (threadIdx.x >> 5);
    const int rowBlock = wid & 127;       // which 16 rows of b
    const int slice    = wid >> 7;        // which K-slice

    // A-fragment addressing (16-bit A 16x32 ISA layout):
    // lane row = rowBlock*16 + (lane&15); lanes 0-15 read K {0..7,16..23},
    // lanes 16-31 read K {8..15,24..31} -> two contiguous 8-float runs.
    const int row  = rowBlock * 16 + (lane & 15);
    const int off0 = (lane < 16) ? 0 : 8;
    const float* xrRow = Xr + (size_t)row * P;
    const float* xiRow = Xi + (size_t)row * P;

    const bf16x16* BrF = reinterpret_cast<const bf16x16*>(Br);
    const bf16x16* BiF = reinterpret_cast<const bf16x16*>(Bi);

    f32x8 accRe[8], accIm[8];
#pragma unroll
    for (int nt = 0; nt < 8; ++nt)
#pragma unroll
        for (int e = 0; e < 8; ++e) { accRe[nt][e] = 0.0f; accIm[nt][e] = 0.0f; }

    const int kc0 = slice * (KSLICE / 32);
    for (int step = 0; step < KSLICE / 32; ++step) {
        const int kc    = kc0 + step;
        const int kBase = kc * 32;

        // Pull the A stream ahead of the WMMA pipeline.
        __builtin_prefetch(xrRow + kBase + 256, 0, 0);
        __builtin_prefetch(xiRow + kBase + 256, 0, 0);

        const float4* pr = reinterpret_cast<const float4*>(xrRow + kBase + off0);
        const float4* pi = reinterpret_cast<const float4*>(xiRow + kBase + off0);
        float4 r0 = pr[0], r1 = pr[1], r2 = pr[4], r3 = pr[5];
        float4 i0 = pi[0], i1 = pi[1], i2 = pi[4], i3 = pi[5];

        bf16x16 aR  = cvt_frag(r0, r1, r2, r3);
        bf16x16 aI  = cvt_frag(i0, i1, i2, i3);
        bf16x16 aIn = neg_frag(aI);

#pragma unroll
        for (int nt = 0; nt < 8; ++nt) {
            const size_t fidx = (size_t)(kc * 8 + nt) * 32 + lane;
            bf16x16 bR = BrF[fidx];     // L2-resident (8.4 MB in 192 MB L2)
            bf16x16 bI = BiF[fidx];
            // re += Xr*Mr - Xi*Mi ; im += Xr*Mi + Xi*Mr
            accRe[nt] = __builtin_amdgcn_wmma_f32_16x16x32_bf16(
                false, aR,  false, bR, (short)0, accRe[nt], false, false);
            accRe[nt] = __builtin_amdgcn_wmma_f32_16x16x32_bf16(
                false, aIn, false, bI, (short)0, accRe[nt], false, false);
            accIm[nt] = __builtin_amdgcn_wmma_f32_16x16x32_bf16(
                false, aR,  false, bI, (short)0, accIm[nt], false, false);
            accIm[nt] = __builtin_amdgcn_wmma_f32_16x16x32_bf16(
                false, aI,  false, bR, (short)0, accIm[nt], false, false);
        }
    }

    // C/D layout: VGPR r holds M = r (lanes 0-15) / 8+r (lanes 16-31), N = lane&15.
    float* pSlice  = partial + (size_t)slice * OUT_FLOATS;
    const int col  = lane & 15;
    const int rAdd = (lane >> 4) * 8;
#pragma unroll
    for (int nt = 0; nt < 8; ++nt) {
#pragma unroll
        for (int r = 0; r < 8; ++r) {
            int b = rowBlock * 16 + rAdd + r;
            int i = nt * 16 + col;
            size_t idx = ((size_t)b * E + i) * 2;
            pSlice[idx + 0] = accRe[nt][r];
            pSlice[idx + 1] = accIm[nt][r];
        }
    }
}

// ------------------------- Phase 3: slice reduction -------------------------
__global__ void qmeas_reduce(const float* __restrict__ partial,
                             float* __restrict__ out) {
    int f = blockIdx.x * blockDim.x + threadIdx.x;
    if (f >= OUT_FLOATS) return;
    float s = 0.0f;
#pragma unroll
    for (int sl = 0; sl < NSLICE; ++sl)
        s += partial[(size_t)sl * OUT_FLOATS + f];
    out[f] = s;   // complex64 interleaved (re, im)
}

// ---------------------------------------------------------------------------
extern "C" void kernel_launch(void* const* d_in, const int* in_sizes, int n_in,
                              void* d_out, int out_size, void* d_ws, size_t ws_size,
                              hipStream_t stream) {
    const float* Xr = (const float*)d_in[0];   // [2048,128,128] f32
    const float* Xi = (const float*)d_in[1];   // [2048,128,128] f32
    const float* Kp = (const float*)d_in[2];   // [128,128,2]    f32
    float* out = (float*)d_out;                // [2048,128] complex64 interleaved
    char*  ws  = (char*)d_ws;

    // ws layout: [0,4MB) Br bf16 | [4MB,8MB) Bi bf16 | [8MB,40MB) partials f32
    __bf16* Br      = (__bf16*)ws;
    __bf16* Bi      = (__bf16*)(ws + (size_t)P * E * sizeof(__bf16));
    float*  partial = (float*)(ws + (size_t)2 * P * E * sizeof(__bf16));

    build_projector_frags<<<(P * E) / 256, 256, 0, stream>>>(Kp, Br, Bi);
    qmeas_gemm<<<(128 * NSLICE) / 8, 256, 0, stream>>>(Xr, Xi, Br, Bi, partial);
    qmeas_reduce<<<OUT_FLOATS / 256, 256, 0, stream>>>(partial, out);
}
